// RNN_54417235641055
// MI455X (gfx1250) — compile-verified, matrix operand order checked by reading, not run on hardware
//
#include <hip/hip_runtime.h>
#include <hip/hip_bf16.h>
#include <math.h>

typedef __attribute__((ext_vector_type(16))) _Float16 v16h;
typedef __attribute__((ext_vector_type(8)))  _Float16 v8h;
typedef __attribute__((ext_vector_type(4)))  _Float16 v4h;
typedef __attribute__((ext_vector_type(8)))  float    v8f;
typedef __attribute__((ext_vector_type(4)))  float    v4f;

#define SEQ  2048
#define BAT  512
#define INP  64
#define HID  128
#define LDSH 200   // halves per staged row: 400B -> 16B-aligned rows, bank-friendly b128 access
#define BUFH (16 * LDSH)

// Branch-free tanh: copysign((1-e)/(1+e), x), e = exp2(-2*log2e*|x|)  (e<=1, no overflow)
__device__ __forceinline__ float tanh_fast(float v) {
    const float ax = __builtin_fabsf(v);
    const float e  = __builtin_amdgcn_exp2f(ax * -2.885390081777927f);
    const float r  = (1.0f - e) * __builtin_amdgcn_rcpf(1.0f + e);
    return __builtin_copysignf(r, v);
}

// Fused Elman RNN scan, transposed WMMA formulation:
//   out^T = [Wh | Wi] @ [h_{t-1} | x_t]^T  ;  h_t = tanh(out + b)
// A = resident weight fragments, B = staged [h|x] tile from LDS, D elements are
// 8 consecutive hidden units per lane -> packed b128 writebacks.
// One workgroup = 16 batch rows for all timesteps; 8 waves = 8 hidden row tiles.
__global__ __launch_bounds__(256)
void rnn_elman_wmma(const float* __restrict__ x,
                    const float* __restrict__ Wh,
                    const float* __restrict__ bh,
                    const float* __restrict__ Wi,
                    const float* __restrict__ bi,
                    float* __restrict__ h_seq,
                    float* __restrict__ h_last)
{
    __shared__ _Float16 Ah[2][BUFH];  // [m][k] f16: k<128 = h_prev, 128..191 = x_t

    const int tid  = threadIdx.x;
    const int lane = tid & 31;
    const int wave = tid >> 5;            // 0..7 -> hidden row tile
    const int b0   = blockIdx.x * 16;     // batch tile base
    const int nl   = lane & 15;
    const int nbase= wave * 16;
    const int n    = nbase + nl;          // hidden unit (A-matrix row) for this lane
    const int koff = (lane >> 4) * 8;     // K sub-offset for A/B fragment halves
    const int mh   = (lane >> 4) * 8;     // D: hidden sub-offset (elements = rows)
    const int nd   = nbase + mh;          // first hidden unit of this lane's D elements

    // ---- preload resident A fragments (f32 -> f16): A[n][k] = k<128 ? Wh[n][k] : Wi[n][k-128]
    v16h Wfrag[6];
#pragma unroll
    for (int c = 0; c < 6; ++c) {
        v16h bf;
#pragma unroll
        for (int h = 0; h < 2; ++h) {
#pragma unroll
            for (int j = 0; j < 8; ++j) {
                const int k = c * 32 + koff + h * 16 + j;
                const float w = (k < HID) ? Wh[n * HID + k] : Wi[n * INP + (k - HID)];
                bf[h * 8 + j] = (_Float16)w;
            }
        }
        Wfrag[c] = bf;
    }
    // per-element bias: D element r corresponds to hidden unit nd + r
    v8f biasv;
#pragma unroll
    for (int r = 0; r < 8; ++r) biasv[r] = bh[nd + r] + bi[nd + r];

    // ---- x loader: each thread owns 4 consecutive floats of the 16x64 tile
    const int xi = tid * 4;
    const int xm = xi >> 6;
    const int xc = xi & 63;
    const float* xptr = x + ((size_t)b0 + xm) * INP + xc;
    const size_t xstep = (size_t)BAT * INP;

    // ---- prologue: zero h region of buf0, stage x_0 into buf0, prefetch x_1
    {
        const int idx = tid * 8;          // 256*8 = 2048 halves = 16x128
        const int m = idx >> 7, col = idx & 127;
        v8h z = {};
        *(v8h*)&Ah[0][m * LDSH + col] = z;

        v4f x0 = __builtin_nontemporal_load((const v4f*)xptr); xptr += xstep;
        v4h xv;
        xv[0] = (_Float16)x0[0]; xv[1] = (_Float16)x0[1];
        xv[2] = (_Float16)x0[2]; xv[3] = (_Float16)x0[3];
        *(v4h*)&Ah[0][xm * LDSH + HID + xc] = xv;
    }
    v4f xnext = __builtin_nontemporal_load((const v4f*)xptr);   // x_1
    xptr += xstep;

    asm volatile("s_wait_dscnt 0x0" ::: "memory");
    asm volatile("s_barrier_signal -1" ::: "memory");

    float hv8[8];                         // carries final hidden for h_last

    for (int t = 0; t < SEQ; ++t) {
        const _Float16* cur = Ah[t & 1];
        _Float16*       nxt = Ah[(t + 1) & 1];

        asm volatile("s_barrier_wait -1" ::: "memory");   // buf[t&1] fully staged

        // load B fragments [h_{t-1} | x_t]^T (identical for all 8 waves)
        v16h Bfrag[6];
#pragma unroll
        for (int c = 0; c < 6; ++c) {
            const _Float16* p = &cur[nl * LDSH + c * 32 + koff];
            v8h lo = *(const v8h*)p;
            v8h hi = *(const v8h*)(p + 16);
            Bfrag[c] = __builtin_shufflevector(lo, hi,
                        0,1,2,3,4,5,6,7,8,9,10,11,12,13,14,15);
        }

        // stage x_{t+1} into the other buffer, issue load of x_{t+2}
        {
            v4h xv;
            xv[0] = (_Float16)xnext[0]; xv[1] = (_Float16)xnext[1];
            xv[2] = (_Float16)xnext[2]; xv[3] = (_Float16)xnext[3];
            *(v4h*)&nxt[xm * LDSH + HID + xc] = xv;
        }
        if (t + 2 < SEQ) { xnext = __builtin_nontemporal_load((const v4f*)xptr); xptr += xstep; }

        // K=192 contraction, two accumulator chains; bias folded into C init
        v8f c0 = biasv;
        v8f c1 = {};
        c0 = __builtin_amdgcn_wmma_f32_16x16x32_f16(false, Wfrag[0], false, Bfrag[0], (short)0, c0, false, false);
        c1 = __builtin_amdgcn_wmma_f32_16x16x32_f16(false, Wfrag[1], false, Bfrag[1], (short)0, c1, false, false);
        c0 = __builtin_amdgcn_wmma_f32_16x16x32_f16(false, Wfrag[2], false, Bfrag[2], (short)0, c0, false, false);
        c1 = __builtin_amdgcn_wmma_f32_16x16x32_f16(false, Wfrag[3], false, Bfrag[3], (short)0, c1, false, false);
        c0 = __builtin_amdgcn_wmma_f32_16x16x32_f16(false, Wfrag[4], false, Bfrag[4], (short)0, c0, false, false);
        c1 = __builtin_amdgcn_wmma_f32_16x16x32_f16(false, Wfrag[5], false, Bfrag[5], (short)0, c1, false, false);

        // branch-free tanh on 8 consecutive hidden units of batch row (b0 + nl)
        v8h hpack;
#pragma unroll
        for (int r = 0; r < 8; ++r) {
            const float hv = tanh_fast(c0[r] + c1[r]);
            hv8[r] = hv;
            hpack[r] = (_Float16)hv;
        }
        // packed f16 feedback into next buffer: row nl, cols nd..nd+7
        *(v8h*)&nxt[nl * LDSH + nd] = hpack;

        asm volatile("s_wait_dscnt 0x0" ::: "memory");
        asm volatile("s_barrier_signal -1" ::: "memory");

        // contiguous f32 output: h_seq[t][b0+nl][nd..nd+7] (never re-read -> non-temporal)
        float* const outp = h_seq + ((size_t)t * BAT + b0 + nl) * HID + nd;
        v4f o0 = {hv8[0], hv8[1], hv8[2], hv8[3]};
        v4f o1 = {hv8[4], hv8[5], hv8[6], hv8[7]};
        __builtin_nontemporal_store(o0, (v4f*)outp);
        __builtin_nontemporal_store(o1, (v4f*)(outp + 4));
    }

    // final hidden state: h_last[b0+nl][nd..nd+7]
    float* const lp = h_last + ((size_t)b0 + nl) * HID + nd;
    v4f l0 = {hv8[0], hv8[1], hv8[2], hv8[3]};
    v4f l1 = {hv8[4], hv8[5], hv8[6], hv8[7]};
    *(v4f*)lp = l0;
    *(v4f*)(lp + 4) = l1;
}

extern "C" void kernel_launch(void* const* d_in, const int* in_sizes, int n_in,
                              void* d_out, int out_size, void* d_ws, size_t ws_size,
                              hipStream_t stream) {
    const float* x  = (const float*)d_in[0];
    const float* Wh = (const float*)d_in[1];
    const float* bh = (const float*)d_in[2];
    const float* Wi = (const float*)d_in[3];
    const float* bi = (const float*)d_in[4];
    float* h_seq  = (float*)d_out;
    float* h_last = h_seq + (size_t)SEQ * BAT * HID;

    rnn_elman_wmma<<<dim3(BAT / 16), dim3(256), 0, stream>>>(
        x, Wh, bh, Wi, bi, h_seq, h_last);
}
